// NSRF_50122268344804
// MI455X (gfx1250) — matrix operand order repaired and checked by reading, not compiled, not run on hardware
//
#include <hip/hip_runtime.h>
#include <math.h>

// ---------------- problem constants ----------------
#define BB 32
#define SS 4
#define QL 12
#define DD 10
#define DL 40
#define VV 32000
#define EE 256
#define HQ 256
#define HD 256
#define HS 512
#define NEGV -1000000000.0f

typedef __attribute__((ext_vector_type(8)))  __bf16 v8bf;
typedef __attribute__((ext_vector_type(16))) __bf16 v16bf;
typedef __attribute__((ext_vector_type(8)))  float  v8f;

// ---------------- device helpers ----------------
__device__ __forceinline__ unsigned short f2bf(float f) {
    union { float f; unsigned u; } v; v.f = f;
    unsigned r = v.u + 0x7fffu + ((v.u >> 16) & 1u);   // round-to-nearest-even
    return (unsigned short)(r >> 16);
}
__device__ __forceinline__ float sigmoidf_(float x) { return 1.0f / (1.0f + __expf(-x)); }
__device__ __forceinline__ float tanh_fast(float x) { return 1.0f - 2.0f / (__expf(2.0f * x) + 1.0f); }

// ---------------- utility kernels ----------------
__global__ void k_fill_f32(float* p, long n, float v) {
    long i = (long)blockIdx.x * blockDim.x + threadIdx.x;
    if (i < n) p[i] = v;
}
__global__ void k_fill_u16(unsigned short* p, long n, unsigned short v) {
    long i = (long)blockIdx.x * blockDim.x + threadIdx.x;
    if (i < n) p[i] = v;
}
__global__ void k_cvt_bf16(const float* __restrict__ s, unsigned short* __restrict__ d, long n) {
    long i = (long)blockIdx.x * blockDim.x + threadIdx.x;
    if (i < n) d[i] = f2bf(s[i]);
}
// concat W1[N,K1] || W2[N,K2] -> bf16 [N,K1+K2]
__global__ void k_wcat(const float* __restrict__ W1, const float* __restrict__ W2,
                       unsigned short* __restrict__ d, int N, int K1, int K2) {
    long i = (long)blockIdx.x * blockDim.x + threadIdx.x;
    int K = K1 + K2;
    if (i >= (long)N * K) return;
    int n = (int)(i / K), k = (int)(i % K);
    float v = (k < K1) ? W1[(long)n * K1 + k] : W2[(long)n * K2 + (k - K1)];
    d[i] = f2bf(v);
}
// embedding gather -> bf16 [nTok, E]
__global__ void k_gather_emb(const int* __restrict__ tok, const float* __restrict__ emb,
                             unsigned short* __restrict__ d, long nTok) {
    long i = (long)blockIdx.x * blockDim.x + threadIdx.x;
    if (i >= nTok * EE) return;
    long t = i / EE; int e = (int)(i % EE);
    d[i] = f2bf(emb[(long)tok[t] * EE + e]);
}

// ---- bf16 WMMA GEMM, 32x32 macro-tile per wave (2 A-frags x 2 B-frags, 4 accumulators) ----
// D[M,N] = act(A[M,K] * W[N,K]^T + bias);  M % 32 == 0, N % 32 == 0, K % 32 == 0
__global__ void __launch_bounds__(256)
k_gemm_bf16(const unsigned short* __restrict__ A, int lda,
            const unsigned short* __restrict__ Bw, int ldb,
            const float* __restrict__ bias,
            float* __restrict__ Dst, int ldd,
            int M, int N, int K, int act)
{
    int wv   = threadIdx.x >> 5;
    int lane = threadIdx.x & 31;
    int Nt = N >> 5;                            // 32-wide N macro-tiles
    int tile = blockIdx.x * (blockDim.x >> 5) + wv;
    int total = (M >> 5) * Nt;
    if (tile >= total) return;                  // wave-uniform: EXEC stays all-ones
    int mt = tile / Nt, nt = tile % Nt;
    int hf = lane >> 4;                         // lane half (0: lanes 0-15, 1: lanes 16-31)
    int r  = lane & 15;

    const __bf16* arow0 = (const __bf16*)A + (size_t)(mt * 32 + r) * lda;
    const __bf16* arow1 = arow0 + (size_t)16 * lda;
    const __bf16* brow0 = (const __bf16*)Bw + (size_t)(nt * 32 + r) * ldb;
    const __bf16* brow1 = brow0 + (size_t)16 * ldb;
    v8f acc00 = {}, acc01 = {}, acc10 = {}, acc11 = {};
    for (int k = 0; k < K; k += 32) {
        // A 16x32 fragments: lanes 0-15 hold K {0..15}, lanes 16-31 {16..31}
        v16bf av0 = *(const v16bf*)(arow0 + k + hf * 16);
        v16bf av1 = *(const v16bf*)(arow1 + k + hf * 16);
        // B 32x16 fragments: W stored [N][K] row-major -> lane r reads K-run [k+hf*16, +16)
        v16bf bv0 = *(const v16bf*)(brow0 + k + hf * 16);
        v16bf bv1 = *(const v16bf*)(brow1 + k + hf * 16);
        acc00 = __builtin_amdgcn_wmma_f32_16x16x32_bf16(false, av0, false, bv0, (short)0, acc00, false, false);
        acc01 = __builtin_amdgcn_wmma_f32_16x16x32_bf16(false, av0, false, bv1, (short)0, acc01, false, false);
        acc10 = __builtin_amdgcn_wmma_f32_16x16x32_bf16(false, av1, false, bv0, (short)0, acc10, false, false);
        acc11 = __builtin_amdgcn_wmma_f32_16x16x32_bf16(false, av1, false, bv1, (short)0, acc11, false, false);
    }
    float b0 = bias ? bias[nt * 32 + r]      : 0.0f;
    float b1 = bias ? bias[nt * 32 + 16 + r] : 0.0f;
    // acc[v] -> (m = mi*16 + v + hf*8, n = nj*16 + r)
    float* d00 = Dst + (size_t)(mt * 32 + hf * 8) * ldd + nt * 32 + r;
#pragma unroll
    for (int v = 0; v < 8; v++) {
        float x00 = acc00[v] + b0, x01 = acc01[v] + b1;
        float x10 = acc10[v] + b0, x11 = acc11[v] + b1;
        if (act) { x00 = tanh_fast(x00); x01 = tanh_fast(x01);
                   x10 = tanh_fast(x10); x11 = tanh_fast(x11); }
        float* dr = d00 + (size_t)v * ldd;
        dr[0]  = x00;
        dr[16] = x01;
        dr[(size_t)16 * ldd]      = x10;
        dr[(size_t)16 * ldd + 16] = x11;
    }
}

// ---------------- encoder step: copy x_t into concat[:,0:E] ----------------
__global__ void k_copy_x(const unsigned short* __restrict__ embb, unsigned short* __restrict__ cc,
                         int M, int t, int T, int ldc) {
    long i = (long)blockIdx.x * blockDim.x + threadIdx.x;
    if (i >= (long)M * EE) return;
    int m = (int)(i / EE), e = (int)(i % EE);
    cc[(size_t)m * ldc + e] = embb[((size_t)m * T + t) * EE + e];
}

// ---------------- masked LSTM gate + running max-pool (encoders) ----------------
__global__ void k_gate_enc(const float* __restrict__ g, float* __restrict__ h, float* __restrict__ c,
                           unsigned short* __restrict__ cc, int ldc, int hoff,
                           float* __restrict__ enc, const int* __restrict__ lens,
                           int t, int M, int H, int dirOff) {
    long i = (long)blockIdx.x * blockDim.x + threadIdx.x;
    if (i >= (long)M * H) return;
    int m = (int)(i / H), j = (int)(i % H);
    const float* gr = g + (size_t)m * 4 * H;
    float hh = h[i], ccv = c[i];
    if (t < lens[m]) {
        float cn = sigmoidf_(gr[H + j]) * ccv + sigmoidf_(gr[j]) * tanh_fast(gr[2 * H + j]);
        float hn = sigmoidf_(gr[3 * H + j]) * tanh_fast(cn);
        ccv = cn; hh = hn;
        float* e = enc + (size_t)m * (2 * H) + dirOff + j;   // fused masked max-pool
        *e = fmaxf(*e, hh);
    }
    h[i] = hh; c[i] = ccv;
    cc[(size_t)m * ldc + hoff + j] = f2bf(hh);               // h for next step's GEMM
}

// ---------------- unmasked LSTM gate (session / decoder) ----------------
__global__ void k_gate_plain(const float* __restrict__ g, float* __restrict__ h, float* __restrict__ c,
                             unsigned short* __restrict__ cc, int ldc, int hoff,
                             float* __restrict__ hs_slot, float* __restrict__ cs_slot,
                             int M, int H) {
    long i = (long)blockIdx.x * blockDim.x + threadIdx.x;
    if (i >= (long)M * H) return;
    int m = (int)(i / H), j = (int)(i % H);
    const float* gr = g + (size_t)m * 4 * H;
    float cn = sigmoidf_(gr[H + j]) * c[i] + sigmoidf_(gr[j]) * tanh_fast(gr[2 * H + j]);
    float hn = sigmoidf_(gr[3 * H + j]) * tanh_fast(cn);
    h[i] = hn; c[i] = cn;
    cc[(size_t)m * ldc + hoff + j] = f2bf(hn);
    if (hs_slot) { hs_slot[i] = hn; cs_slot[i] = cn; }
}

// ---------------- session concat build: [enc_q(b,s) | h_sess(b)] -> bf16 [32,1024] ----------------
__global__ void k_concat_sess(const float* __restrict__ encq, int s, const float* __restrict__ hs,
                              unsigned short* __restrict__ cc) {
    long i = (long)blockIdx.x * blockDim.x + threadIdx.x;
    if (i >= (long)BB * 1024) return;
    int b = (int)(i / 1024), j = (int)(i % 1024);
    float v = (j < 512) ? encq[((size_t)b * SS + s) * 512 + j] : hs[(size_t)b * 512 + (j - 512)];
    cc[i] = f2bf(v);
}

// ---------------- score + BCE accumulation ----------------
__global__ void k_score_bce(const float* __restrict__ comb, const float* __restrict__ encd, int s,
                            const float* __restrict__ lab, float* __restrict__ accum) {
    int i = blockIdx.x * blockDim.x + threadIdx.x;
    if (i >= BB * DD) return;
    int b = i / DD, d = i % DD;
    const float* cb = comb + (size_t)b * 512;
    const float* ed = encd + ((size_t)(b * SS + s) * DD + d) * 512;
    float sc = 0.f;
    for (int k = 0; k < 512; k++) sc += cb[k] * ed[k];
    float l = lab[(b * SS + s) * DD + d];
    float bce = fmaxf(sc, 0.f) - sc * l + logf(1.f + __expf(-fabsf(sc)));
    atomicAdd(&accum[0], bce);
}

// ---------------- decoder init: dec_h/dec_c from hs/cs stores, seed concat h-part ----------------
__global__ void k_dec_init(const float* __restrict__ hss, const float* __restrict__ css,
                           float* __restrict__ dh, float* __restrict__ dc,
                           unsigned short* __restrict__ cc) {
    long i = (long)blockIdx.x * blockDim.x + threadIdx.x;
    if (i >= 96L * 512) return;
    int n = (int)(i / 512), j = (int)(i % 512);
    int b = n / 3, s = n % 3;                        // sessions s = 0..2 feed target s+1
    float hv = hss[((size_t)s * BB + b) * 512 + j];
    dh[i] = hv;
    dc[i] = css[((size_t)s * BB + b) * 512 + j];
    cc[(size_t)n * 768 + 256 + j] = f2bf(hv);
}

// ---------------- decoder concat x-part: emb of query (b, s+1, t) ----------------
__global__ void k_dec_x(const unsigned short* __restrict__ embq, unsigned short* __restrict__ cc, int t) {
    long i = (long)blockIdx.x * blockDim.x + threadIdx.x;
    if (i >= 96L * EE) return;
    int n = (int)(i / EE), e = (int)(i % EE);
    int b = n / 3, s = n % 3;
    long row = (long)(b * SS + (s + 1)) * QL + t;
    cc[(size_t)n * 768 + e] = embq[row * EE + e];
}

// ---------------- log-softmax NLL over V=32000, block per row ----------------
__global__ void k_nll(const float* __restrict__ logits, const int* __restrict__ sq,
                      const int* __restrict__ sql, int t, float* __restrict__ accum) {
    __shared__ float red[256];
    int n = blockIdx.x;                              // 0..95
    int b = n / 3, s = n % 3;
    const float* row = logits + (size_t)n * VV;
    float mx = -INFINITY;
    for (int i = threadIdx.x; i < VV; i += 256) mx = fmaxf(mx, row[i]);
    red[threadIdx.x] = mx; __syncthreads();
    for (int st = 128; st > 0; st >>= 1) {
        if (threadIdx.x < st) red[threadIdx.x] = fmaxf(red[threadIdx.x], red[threadIdx.x + st]);
        __syncthreads();
    }
    mx = red[0]; __syncthreads();
    float se = 0.f;
    for (int i = threadIdx.x; i < VV; i += 256) se += __expf(row[i] - mx);
    red[threadIdx.x] = se; __syncthreads();
    for (int st = 128; st > 0; st >>= 1) {
        if (threadIdx.x < st) red[threadIdx.x] += red[threadIdx.x + st];
        __syncthreads();
    }
    if (threadIdx.x == 0) {
        int len = sql[b * SS + s + 1];
        if (t < len) {
            int tgt = sq[(long)(b * SS + (s + 1)) * QL + (t + 1)];
            float nll = -(row[tgt] - mx - logf(red[0]));
            atomicAdd(&accum[1], nll);
            atomicAdd(&accum[2], 1.0f);
        }
    }
}

__global__ void k_final(const float* __restrict__ accum, float* __restrict__ out) {
    float click = accum[0] / (float)(SS * BB * DD);
    float dec = (accum[2] > 0.f) ? accum[1] / accum[2] : 0.f;
    out[0] = click + dec;
}

// ---------------- host orchestration ----------------
extern "C" void kernel_launch(void* const* d_in, const int* in_sizes, int n_in,
                              void* d_out, int out_size, void* d_ws, size_t ws_size,
                              hipStream_t stream) {
    (void)in_sizes; (void)n_in; (void)out_size; (void)ws_size;
    const int*   sq   = (const int*)d_in[0];
    const int*   sql  = (const int*)d_in[1];
    const int*   rd   = (const int*)d_in[2];
    const int*   rdl  = (const int*)d_in[3];
    const float* lab  = (const float*)d_in[4];
    const float* emb  = (const float*)d_in[5];
    const float* qWihf = (const float*)d_in[6],  *qWhhf = (const float*)d_in[7],  *qbf = (const float*)d_in[8];
    const float* qWihb = (const float*)d_in[9],  *qWhhb = (const float*)d_in[10], *qbb = (const float*)d_in[11];
    const float* dWihf = (const float*)d_in[12], *dWhhf = (const float*)d_in[13], *dbf = (const float*)d_in[14];
    const float* dWihb = (const float*)d_in[15], *dWhhb = (const float*)d_in[16], *dbb = (const float*)d_in[17];
    const float* sWih  = (const float*)d_in[18], *sWhh  = (const float*)d_in[19], *sb  = (const float*)d_in[20];
    const float* projW = (const float*)d_in[21], *projb = (const float*)d_in[22];
    const float* decWih = (const float*)d_in[23], *decWhh = (const float*)d_in[24], *decb = (const float*)d_in[25];
    const float* outW  = (const float*)d_in[26], *outb  = (const float*)d_in[27];

    // bump allocator over workspace
    char* base = (char*)d_ws; size_t off = 0;
    auto alloc = [&](size_t bytes) -> void* {
        void* p = base + off; off += (bytes + 255) & ~(size_t)255; return p;
    };
    unsigned short* qWcatf = (unsigned short*)alloc(1024L * 512 * 2);
    unsigned short* qWcatb = (unsigned short*)alloc(1024L * 512 * 2);
    unsigned short* dWcatf = (unsigned short*)alloc(1024L * 512 * 2);
    unsigned short* dWcatb = (unsigned short*)alloc(1024L * 512 * 2);
    unsigned short* sWcat  = (unsigned short*)alloc(2048L * 1024 * 2);
    unsigned short* projWb = (unsigned short*)alloc(512L * 1024 * 2);
    unsigned short* decWcat = (unsigned short*)alloc(2048L * 768 * 2);
    unsigned short* outWb  = (unsigned short*)alloc((size_t)VV * 512 * 2);
    unsigned short* embq_b = (unsigned short*)alloc((size_t)BB * SS * QL * EE * 2);
    unsigned short* embd_b = (unsigned short*)alloc((size_t)BB * SS * DD * DL * EE * 2);
    unsigned short* cc_enc = (unsigned short*)alloc(1280L * 512 * 2);
    unsigned short* cc_s   = (unsigned short*)alloc(32L * 1024 * 2);
    unsigned short* cc_dec = (unsigned short*)alloc(96L * 768 * 2);
    float* gbuf   = (float*)alloc(1280L * 1024 * 4);
    float* h_enc  = (float*)alloc(1280L * 256 * 4);
    float* c_enc  = (float*)alloc(1280L * 256 * 4);
    float* encq   = (float*)alloc(128L * 512 * 4);
    float* encd   = (float*)alloc(1280L * 512 * 4);
    float* h_sess = (float*)alloc(32L * 512 * 4);
    float* c_sess = (float*)alloc(32L * 512 * 4);
    float* hs_st  = (float*)alloc((size_t)SS * 32 * 512 * 4);
    float* cs_st  = (float*)alloc((size_t)SS * 32 * 512 * 4);
    float* comb   = (float*)alloc(32L * 512 * 4);
    float* dec_h  = (float*)alloc(96L * 512 * 4);
    float* dec_c  = (float*)alloc(96L * 512 * 4);
    float* logits = (float*)alloc(96L * (size_t)VV * 4);
    float* accum  = (float*)alloc(4 * 4);

    auto gridE = [](long n) { return (unsigned)((n + 255) / 256); };
    auto gemm = [&](const unsigned short* A, int lda, const unsigned short* W, int ldb,
                    const float* bias, float* Dst, int ldd, int M, int N, int K, int act) {
        long tiles = (long)(M / 32) * (N / 32);
        k_gemm_bf16<<<(unsigned)((tiles + 7) / 8), 256, 0, stream>>>(A, lda, W, ldb, bias, Dst, ldd, M, N, K, act);
    };

    // --- prep: zero accumulators, build bf16 weights + embeddings ---
    k_fill_f32<<<1, 4, 0, stream>>>(accum, 4, 0.0f);
    k_wcat<<<gridE(1024L * 512), 256, 0, stream>>>(qWihf, qWhhf, qWcatf, 1024, 256, 256);
    k_wcat<<<gridE(1024L * 512), 256, 0, stream>>>(qWihb, qWhhb, qWcatb, 1024, 256, 256);
    k_wcat<<<gridE(1024L * 512), 256, 0, stream>>>(dWihf, dWhhf, dWcatf, 1024, 256, 256);
    k_wcat<<<gridE(1024L * 512), 256, 0, stream>>>(dWihb, dWhhb, dWcatb, 1024, 256, 256);
    k_wcat<<<gridE(2048L * 1024), 256, 0, stream>>>(sWih, sWhh, sWcat, 2048, 512, 512);
    k_wcat<<<gridE(2048L * 768), 256, 0, stream>>>(decWih, decWhh, decWcat, 2048, 256, 512);
    k_cvt_bf16<<<gridE(512L * 1024), 256, 0, stream>>>(projW, projWb, 512L * 1024);
    k_cvt_bf16<<<gridE((long)VV * 512), 256, 0, stream>>>(outW, outWb, (long)VV * 512);
    k_gather_emb<<<gridE((long)BB * SS * QL * EE), 256, 0, stream>>>(sq, emb, embq_b, (long)BB * SS * QL);
    k_gather_emb<<<gridE((long)BB * SS * DD * DL * EE), 256, 0, stream>>>(rd, emb, embd_b, (long)BB * SS * DD * DL);

    // --- BiLSTM encoder (generic over query / doc) ---
    auto run_encoder = [&](const unsigned short* embb, int M, int T, const int* lens,
                           float* enc, const unsigned short* Wf, const float* bf_,
                           const unsigned short* Wb, const float* bb_) {
        k_fill_f32<<<gridE((long)M * 512), 256, 0, stream>>>(enc, (long)M * 512, NEGV);
        for (int dir = 0; dir < 2; dir++) {
            const unsigned short* W = dir ? Wb : Wf;
            const float* bias = dir ? bb_ : bf_;
            int dirOff = dir ? 256 : 0;
            k_fill_f32<<<gridE((long)M * 256), 256, 0, stream>>>(h_enc, (long)M * 256, 0.0f);
            k_fill_f32<<<gridE((long)M * 256), 256, 0, stream>>>(c_enc, (long)M * 256, 0.0f);
            k_fill_u16<<<gridE((long)M * 512), 256, 0, stream>>>(cc_enc, (long)M * 512, 0);
            for (int step = 0; step < T; step++) {
                int t = dir ? (T - 1 - step) : step;
                k_copy_x<<<gridE((long)M * EE), 256, 0, stream>>>(embb, cc_enc, M, t, T, 512);
                gemm(cc_enc, 512, W, 512, bias, gbuf, 1024, M, 1024, 512, 0);
                k_gate_enc<<<gridE((long)M * 256), 256, 0, stream>>>(
                    gbuf, h_enc, c_enc, cc_enc, 512, 256, enc, lens, t, M, 256, dirOff);
            }
        }
    };
    run_encoder(embq_b, BB * SS, QL, sql, encq, qWcatf, qbf, qWcatb, qbb);
    run_encoder(embd_b, BB * SS * DD, DL, rdl, encd, dWcatf, dbf, dWcatb, dbb);

    // --- session loop ---
    k_fill_f32<<<gridE(32L * 512), 256, 0, stream>>>(h_sess, 32L * 512, 0.0f);
    k_fill_f32<<<gridE(32L * 512), 256, 0, stream>>>(c_sess, 32L * 512, 0.0f);
    for (int s = 0; s < SS; s++) {
        k_concat_sess<<<gridE(32L * 1024), 256, 0, stream>>>(encq, s, h_sess, cc_s);
        gemm(cc_s, 1024, projWb, 1024, projb, comb, 512, 32, 512, 1024, 1);   // tanh fused
        k_score_bce<<<gridE(BB * DD), 256, 0, stream>>>(comb, encd, s, lab, accum);
        gemm(cc_s, 1024, sWcat, 1024, sb, gbuf, 2048, 32, 2048, 1024, 0);
        k_gate_plain<<<gridE(32L * 512), 256, 0, stream>>>(
            gbuf, h_sess, c_sess, cc_s, 1024, 512,
            hs_st + (size_t)s * 32 * 512, cs_st + (size_t)s * 32 * 512, 32, 512);
    }

    // --- decoder ---
    k_dec_init<<<gridE(96L * 512), 256, 0, stream>>>(hs_st, cs_st, dec_h, dec_c, cc_dec);
    for (int t = 0; t < QL - 1; t++) {
        k_dec_x<<<gridE(96L * EE), 256, 0, stream>>>(embq_b, cc_dec, t);
        gemm(cc_dec, 768, decWcat, 768, decb, gbuf, 2048, 96, 2048, 768, 0);
        k_gate_plain<<<gridE(96L * 512), 256, 0, stream>>>(
            gbuf, dec_h, dec_c, cc_dec, 768, 256, nullptr, nullptr, 96, 512);
        // logits: A = bf16 h stored inside cc_dec[:,256:768] (lda=768)
        gemm(cc_dec + 256, 768, outWb, 512, outb, logits, VV, 96, VV, 512, 0);
        k_nll<<<96, 256, 0, stream>>>(logits, sq, sql, t, accum);
    }

    k_final<<<1, 1, 0, stream>>>(accum, (float*)d_out);
}
